// AverageAttentionTransformer_63754494542069
// MI455X (gfx1250) — matrix operand order repaired and checked by reading, not compiled, not run on hardware
//
#include <hip/hip_runtime.h>
#include <math.h>

typedef __attribute__((ext_vector_type(16))) _Float16 v16h;
typedef __attribute__((ext_vector_type(8)))  float    v8f;

#define D    64
#define SQ   1000
#define SP   1008      // padded to 63*16
#define XST  66        // f32 LDS row stride (bank-friendly)
#define HST  66        // f16 staging row stride
#define NL   4
#define NT   256       // threads per block (8 wave32)
#define NW   8

struct alignas(64) Lds {
  float     X[SP * XST];            // 266112 B  activation tensor, f32 resident
  _Float16  Hs[NW][16 * HST];       //  16896 B  per-wave h staging (layout bounce)
  _Float16  Bfrag[2][8][512];       //  16384 B  f16 B-fragments: [ff1|ff2][Kblk*4+j][lane*16+i]
  float     part[4][D];
  float     avg[D];
  float     attn[D];
  float     gS[D], bS[D];
  float     b1S[D], b2S[D];
};

// f16 A-fragment (16x32) from an f32 LDS row. lane<16: M=lane;
// element i: K = koff + 16*(i/8) + 8*half + i%8.  Pairs contiguous -> b64 loads.
__device__ inline v16h load_a_f32(const float* row, int koff, int half) {
  v16h a;
#pragma unroll
  for (int t = 0; t < 8; ++t) {
    int kb = koff + ((t >= 4) ? 16 : 0) + half * 8 + (t & 3) * 2;
    float2 f = *reinterpret_cast<const float2*>(row + kb);
    a[2 * t]     = (_Float16)f.x;
    a[2 * t + 1] = (_Float16)f.y;
  }
  return a;
}

__device__ inline v16h load_a_f16(const _Float16* row, int koff, int half) {
  v16h a;
#pragma unroll
  for (int t = 0; t < 8; ++t) {
    int kb = koff + ((t >= 4) ? 16 : 0) + half * 8 + (t & 3) * 2;
    a[2 * t]     = row[kb];
    a[2 * t + 1] = row[kb + 1];
  }
  return a;
}

// LayerNorm all SP rows in place; optionally add lds->attn broadcast first.
template <bool ADD>
__device__ inline void ln_pass(Lds* lds, int tid) {
  for (int r = tid; r < SP; r += NT) {
    float* row = &lds->X[r * XST];
    float xv[D];
    float mu = 0.f;
#pragma unroll
    for (int k = 0; k < D; ++k) {
      float v = row[k];
      if (ADD) v += lds->attn[k];
      xv[k] = v;
      mu += v;
    }
    mu *= (1.f / D);
    float var = 0.f;
#pragma unroll
    for (int k = 0; k < D; ++k) { float d = xv[k] - mu; var += d * d; }
    var *= (1.f / D);
    float rs = rsqrtf(var + 1e-5f);
#pragma unroll
    for (int k = 0; k < D; ++k)
      row[k] = (xv[k] - mu) * rs * lds->gS[k] + lds->bS[k];
  }
}

// One 16-row FFN tile: x -> relu(x@W1^T+b1) -> @W2^T+b2, accumulated into X.
__device__ inline void ffn_tile(Lds* lds, int m, int w, int lane, int half, int ln16,
                                const float* bb1, const float* bb2) {
  const float* xrow = &lds->X[(m * 16 + ln16) * XST];
  v16h a0 = load_a_f32(xrow, 0,  half);
  v16h a1 = load_a_f32(xrow, 32, half);
#pragma unroll
  for (int j = 0; j < 4; ++j) {
    v16h bk0 = *reinterpret_cast<const v16h*>(&lds->Bfrag[0][0 + j][lane * 16]);
    v16h bk1 = *reinterpret_cast<const v16h*>(&lds->Bfrag[0][4 + j][lane * 16]);
    v8f c = {};
    c = __builtin_amdgcn_wmma_f32_16x16x32_f16(false, a0, false, bk0, (short)0, c, false, false);
    c = __builtin_amdgcn_wmma_f32_16x16x32_f16(false, a1, false, bk1, (short)0, c, false, false);
#pragma unroll
    for (int v = 0; v < 8; ++v) {
      float hv = c[v] + bb1[j];
      hv = hv > 0.f ? hv : 0.f;
      lds->Hs[w][(v + half * 8) * HST + j * 16 + ln16] = (_Float16)hv;
    }
  }
  const _Float16* hrow = &lds->Hs[w][ln16 * HST];
  v16h h0 = load_a_f16(hrow, 0,  half);
  v16h h1 = load_a_f16(hrow, 32, half);
#pragma unroll
  for (int j = 0; j < 4; ++j) {
    v16h bk0 = *reinterpret_cast<const v16h*>(&lds->Bfrag[1][0 + j][lane * 16]);
    v16h bk1 = *reinterpret_cast<const v16h*>(&lds->Bfrag[1][4 + j][lane * 16]);
    v8f c = {};
    c = __builtin_amdgcn_wmma_f32_16x16x32_f16(false, h0, false, bk0, (short)0, c, false, false);
    c = __builtin_amdgcn_wmma_f32_16x16x32_f16(false, h1, false, bk1, (short)0, c, false, false);
#pragma unroll
    for (int v = 0; v < 8; ++v) {
      int r = m * 16 + v + half * 8;
      float* p = &lds->X[r * XST + j * 16 + ln16];
      *p = *p + c[v] + bb2[j];
    }
  }
}

__global__ __launch_bounds__(NT) void avg_attn_transformer_kernel(
    const int*   __restrict__ tokens,
    const float* __restrict__ emb,
    const float* __restrict__ lin_w, const float* __restrict__ lin_b,
    const float* __restrict__ ff1_w, const float* __restrict__ ff1_b,
    const float* __restrict__ ff2_w, const float* __restrict__ ff2_b,
    const float* __restrict__ n1_g,  const float* __restrict__ n1_b,
    const float* __restrict__ n2_g,  const float* __restrict__ n2_b,
    const float* __restrict__ out_w, const float* __restrict__ out_b,
    float* __restrict__ out)
{
  __shared__ Lds lds;
  const int tid  = threadIdx.x;
  const int b    = blockIdx.x;
  const int lane = tid & 31;
  const int w    = tid >> 5;
  const int half = lane >> 4;
  const int ln16 = lane & 15;

  // ---------------- embed + positional encoding -> LDS ----------------
  for (int s = tid; s < SP; s += NT) {
    float* row = &lds.X[s * XST];
    if (s < SQ) {
      int tok = tokens[(long)b * SQ + s];
      const float* e = emb + (long)tok * D;
#pragma unroll
      for (int i = 0; i < D / 2; ++i) {
        float div = expf(-(float)(2 * i) * (9.210340371976184f / 64.f)); // ln(1e4)/d
        float ang = (float)s * div;
        row[2 * i]     = e[2 * i]     * 8.f + sinf(ang);
        row[2 * i + 1] = e[2 * i + 1] * 8.f + cosf(ang);
      }
    } else {
#pragma unroll
      for (int k = 0; k < D; ++k) row[k] = 0.f;
    }
  }
  __syncthreads();

  for (int l = 0; l < NL; ++l) {
    // ---------------- mean over sequence ----------------
    {
      int k = tid & 63, g = tid >> 6;
      float acc = 0.f;
      for (int s = g; s < SQ; s += 4) acc += lds.X[s * XST + k];
      lds.part[g][k] = acc;
    }
    __syncthreads();
    if (tid < D)
      lds.avg[tid] = (lds.part[0][tid] + lds.part[1][tid] +
                      lds.part[2][tid] + lds.part[3][tid]) * (1.f / SQ);
    __syncthreads();
    // ---------------- attn matvec + LN1 params ----------------
    if (tid < D) {
      const float* wr = lin_w + ((long)l * D + tid) * D;
      float a = lin_b[l * D + tid];
#pragma unroll
      for (int d2 = 0; d2 < D; ++d2) a += lds.avg[d2] * wr[d2];
      lds.attn[tid] = a;
    } else if (tid < 2 * D) {
      lds.gS[tid - D] = n1_g[l * D + (tid - D)];
    } else if (tid < 3 * D) {
      lds.bS[tid - 2 * D] = n1_b[l * D + (tid - 2 * D)];
    }
    __syncthreads();
    ln_pass<true>(&lds, tid);          // x = LN(x + attn)
    __syncthreads();

    // ---------------- pack FF weights as f16 B-fragments + biases + LN2 params ----
    for (int e = tid; e < 2 * 8 * 512; e += NT) {
      int mat  = e >> 12;
      int rem  = e & 4095;
      int frag = rem >> 9;
      int li   = rem & 511;
      int n = li >> 4, i = li & 15;
      int Kblk = frag >> 2, j = frag & 3;
      int K = Kblk * 32 + ((i >> 3) << 4) + ((n >> 4) << 3) + (i & 7);   // B layout: table order, lane-half picks K sub-block of 8
      int N = j * 16 + (n & 15);
      const float* W = (mat == 0 ? ff1_w : ff2_w) + (long)l * D * D;
      lds.Bfrag[mat][frag][li] = (_Float16)W[N * D + K];
    }
    if      (tid < D)     lds.b1S[tid]         = ff1_b[l * D + tid];
    else if (tid < 2 * D) lds.b2S[tid - D]     = ff2_b[l * D + (tid - D)];
    else if (tid < 3 * D) lds.gS[tid - 2 * D]  = n2_g[l * D + (tid - 2 * D)];
    else                  lds.bS[tid - 3 * D]  = n2_b[l * D + (tid - 3 * D)];
    __syncthreads();

    // per-lane bias registers (invariant over tiles)
    float bb1[4], bb2[4];
#pragma unroll
    for (int j = 0; j < 4; ++j) {
      bb1[j] = lds.b1S[j * 16 + ln16];
      bb2[j] = lds.b2S[j * 16 + ln16];
    }

    // ---------------- FFN via WMMA: each wave owns 16-row tiles ----------------
    // unroll 2 tiles to give the scheduler independent wmma/ds chains
    int m = w;
    for (; m + NW < SP / 16; m += 2 * NW) {
      ffn_tile(&lds, m,      w, lane, half, ln16, bb1, bb2);
      ffn_tile(&lds, m + NW, w, lane, half, ln16, bb1, bb2);
    }
    if (m < SP / 16)
      ffn_tile(&lds, m, w, lane, half, ln16, bb1, bb2);
    __syncthreads();
    ln_pass<false>(&lds, tid);         // x = LN(x + ffn)
    __syncthreads();
  }

  // ---------------- output head: logits[s][0..1] ----------------
  if      (tid < D)     lds.gS[tid]     = out_w[tid];
  else if (tid < 2 * D) lds.bS[tid - D] = out_w[D + (tid - D)];
  __syncthreads();
  float ob0 = out_b[0], ob1 = out_b[1];
  for (int s = tid; s < SQ; s += NT) {
    const float* row = &lds.X[s * XST];
    float a0 = ob0, a1 = ob1;
#pragma unroll
    for (int k = 0; k < D; ++k) {
      a0 += row[k] * lds.gS[k];
      a1 += row[k] * lds.bS[k];
    }
    long o = ((long)b * SQ + s) * 2;
    out[o]     = a0;
    out[o + 1] = a1;
  }
}

extern "C" void kernel_launch(void* const* d_in, const int* in_sizes, int n_in,
                              void* d_out, int out_size, void* d_ws, size_t ws_size,
                              hipStream_t stream) {
  (void)n_in; (void)out_size; (void)d_ws; (void)ws_size;
  const int*   tokens = (const int*)  d_in[0];
  const float* emb    = (const float*)d_in[1];
  const float* lin_w  = (const float*)d_in[2];
  const float* lin_b  = (const float*)d_in[3];
  const float* ff1_w  = (const float*)d_in[4];
  const float* ff1_b  = (const float*)d_in[5];
  const float* ff2_w  = (const float*)d_in[6];
  const float* ff2_b  = (const float*)d_in[7];
  const float* n1_g   = (const float*)d_in[8];
  const float* n1_b   = (const float*)d_in[9];
  const float* n2_g   = (const float*)d_in[10];
  const float* n2_b   = (const float*)d_in[11];
  const float* out_w  = (const float*)d_in[12];
  const float* out_b  = (const float*)d_in[13];
  float* out = (float*)d_out;

  int nb = in_sizes[0] / SQ;   // batch = 512
  avg_attn_transformer_kernel<<<nb, NT, 0, stream>>>(
      tokens, emb, lin_w, lin_b, ff1_w, ff1_b, ff2_w, ff2_b,
      n1_g, n1_b, n2_g, n2_b, out_w, out_b, out);
}
